// DSHLoss_55654186221915
// MI455X (gfx1250) — compile-verified, matrix operand order checked by reading, not run on hardware
//
#include <hip/hip_runtime.h>

// ---------------- problem constants (from reference) ----------------
#define BIT      64
#define NCLS     100
#define NPAD     128             // y padded K for bf16 WMMA (100 -> 128)
#define NTRAIN   100000
#define BATCH    512
#define MARGIN   128.0f          // m = 2*bit
#define ALPHA    0.1f

#define NTILES   (NTRAIN / 16)   // 6250 N-tiles of 16 rows
#define WPB      4               // waves per block (128 threads, wave32)
#define NBLOCKS  ((NTILES + WPB - 1) / WPB)   // 1563

typedef __attribute__((ext_vector_type(2)))  float   v2f;
typedef __attribute__((ext_vector_type(4)))  float   v4f;
typedef __attribute__((ext_vector_type(8)))  float   v8f;
typedef __attribute__((ext_vector_type(8)))  __bf16  v8bf;
typedef __attribute__((ext_vector_type(16))) __bf16  v16bf;

// ---- workspace layout (bytes) ----
#define WS_OVR      0                    // int[100000]           400000 B
#define WS_USQ      400000               // float[512]              2048 B
#define WS_UREORD   402048               // float[512*64]         131072 B
#define WS_YBF      533120               // __bf16[512*128]       131072 B
#define WS_PART     664192               // float[NBLOCKS]

// Pack one K-chunk (32 wide, padded past NCLS with zeros) of an f32 row into
// the bf16 A/B per-lane WMMA layout (used only in once-per-wave B setup):
//   vec elems 2j,2j+1   <- K = kbase + 8*hf + 2j (+1)
//   vec elems 8+2j,..+1 <- K = kbase + 16 + 8*hf + 2j (+1)
__device__ inline v16bf pack_bf16_chunk(const float* __restrict__ row, int kbase, int hf) {
    v16bf v;
#pragma unroll
    for (int j = 0; j < 4; ++j) {
        int k0 = kbase + 8 * hf + 2 * j;
        int k1 = kbase + 16 + 8 * hf + 2 * j;
        float a0 = row[(k0     < NCLS) ? k0     : (NCLS - 1)]; a0 = (k0     < NCLS) ? a0 : 0.0f;
        float a1 = row[(k0 + 1 < NCLS) ? k0 + 1 : (NCLS - 1)]; a1 = (k0 + 1 < NCLS) ? a1 : 0.0f;
        float b0 = row[(k1     < NCLS) ? k1     : (NCLS - 1)]; b0 = (k1     < NCLS) ? b0 : 0.0f;
        float b1 = row[(k1 + 1 < NCLS) ? k1 + 1 : (NCLS - 1)]; b1 = (k1 + 1 < NCLS) ? b1 : 0.0f;
        v[2 * j]         = (__bf16)a0;
        v[2 * j + 1]     = (__bf16)a1;
        v[8 + 2 * j]     = (__bf16)b0;
        v[8 + 2 * j + 1] = (__bf16)b1;
    }
    return v;
}

// ---------------- override map: ovr[n] = batch row replacing U/Y row n, else -1
__global__ void dsh_init_ovr(int* __restrict__ ovr) {
    int i = blockIdx.x * 256 + threadIdx.x;
    if (i < NTRAIN) ovr[i] = -1;
}

__global__ void dsh_set_ovr(int* __restrict__ ovr, const int* __restrict__ ind) {
    int b = blockIdx.x * 256 + threadIdx.x;
    if (b < BATCH) {
        int t = ind[b];
        if (t >= 0 && t < NTRAIN) ovr[t] = b;
    }
}

// ---------------- per-batch-row precompute: usq, u reordered, y in bf16 ----
// u_reord row layout: [pairs k={4s,4s+1}, s=0..15][pairs k={4s+2,4s+3}]
// so each lane-half's 16 WMMA f32 K-pairs are 32 contiguous floats.
__global__ __launch_bounds__(256) void dsh_pack_rows(
    const float* __restrict__ u, const float* __restrict__ y,
    float* __restrict__ usq, float* __restrict__ u_reord,
    __bf16* __restrict__ ybf)
{
    int t = blockIdx.x * 256 + threadIdx.x;
    if (t >= BATCH) return;

    const float* ur = u + (size_t)t * BIT;
    float*      dst = u_reord + (size_t)t * BIT;
    float s = 0.0f;
#pragma unroll
    for (int p = 0; p < 16; ++p) {
        float a0 = ur[4 * p + 0], a1 = ur[4 * p + 1];
        float a2 = ur[4 * p + 2], a3 = ur[4 * p + 3];
        s += a0 * a0 + a1 * a1 + a2 * a2 + a3 * a3;
        dst[2 * p]          = a0;  dst[2 * p + 1]      = a1;
        dst[32 + 2 * p]     = a2;  dst[32 + 2 * p + 1] = a3;
    }
    usq[t] = s;

    const float* yr = y + (size_t)t * NCLS;
    __bf16*      yd = ybf + (size_t)t * NPAD;
#pragma unroll
    for (int c = 0; c < NPAD; ++c)
        yd[c] = (c < NCLS) ? (__bf16)yr[c] : (__bf16)0.0f;
}

// ---------------- main: per-wave 16-col N tile, loop over 32 batch tiles ----
__global__ __launch_bounds__(128) void dsh_main(
    const float* __restrict__ u, const float* __restrict__ y,
    const float* __restrict__ Ubuf, const float* __restrict__ Ybuf,
    const int* __restrict__ ovr,
    const float* __restrict__ usq, const float* __restrict__ u_reord,
    const __bf16* __restrict__ ybf,
    float* __restrict__ partials)
{
    const int lane = threadIdx.x & 31;
    const int wave = threadIdx.x >> 5;
    const int nt   = blockIdx.x * WPB + wave;
    const int idx  = lane & 15;   // N column within tile / A row within tile
    const int hf   = lane >> 4;   // lane half selects K sub-pairs

    float lane_loss = 0.0f;

    if (nt < NTILES) {
        const int n_global = nt * 16 + idx;
        const int o = ovr[n_global];
        const float* Urow = (o >= 0) ? (u + (size_t)o * BIT)  : (Ubuf + (size_t)n_global * BIT);
        const float* Yrow = (o >= 0) ? (y + (size_t)o * NCLS) : (Ybuf + (size_t)n_global * NCLS);

        // B operand for u-GEMM: Ut tile, f32 16x16x4 layout. Also |U_n|^2.
        v2f  bu[16];
        float Usq_p = 0.0f;
#pragma unroll
        for (int s = 0; s < 16; ++s) {
            v2f t = *(const v2f*)(Urow + 4 * s + 2 * hf);
            bu[s] = t;
            Usq_p += t[0] * t[0] + t[1] * t[1];
        }
        const float Usq_n = Usq_p + __shfl_xor(Usq_p, 16, 32);  // full-row norm^2

        // B operand for y-GEMM: Yt tile, bf16 16x16x32 layout, K padded to 128.
        v16bf by[4];
#pragma unroll
        for (int c = 0; c < 4; ++c) by[c] = pack_bf16_chunk(Yrow, 32 * c, hf);

        for (int bt = 0; bt < BATCH / 16; ++bt) {
            // A operand for u-GEMM: 32 contiguous floats = 8 x b128 loads
            const float* urow = u_reord + (size_t)(bt * 16 + idx) * BIT + hf * 32;
            v2f au[16];
#pragma unroll
            for (int q = 0; q < 8; ++q) {
                v4f t = *(const v4f*)(urow + 4 * q);
                au[2 * q][0]     = t[0];  au[2 * q][1]     = t[1];
                au[2 * q + 1][0] = t[2];  au[2 * q + 1][1] = t[3];
            }

            // A operand for y-GEMM: pre-packed bf16, 2 x b128 loads per chunk
            const __bf16* yrow = ybf + (size_t)(bt * 16 + idx) * NPAD;
            v16bf ay[4];
#pragma unroll
            for (int c = 0; c < 4; ++c) {
                v8bf lo = *(const v8bf*)(yrow + 32 * c + 8 * hf);
                v8bf hi = *(const v8bf*)(yrow + 32 * c + 16 + 8 * hf);
                ay[c] = __builtin_shufflevector(lo, hi,
                         0, 1, 2, 3, 4, 5, 6, 7, 8, 9, 10, 11, 12, 13, 14, 15);
            }

            // |u_m|^2 for the 8 output rows this lane holds (broadcast loads)
            const float* uq = usq + bt * 16 + 8 * hf;
            v4f q0 = *(const v4f*)(uq);
            v4f q1 = *(const v4f*)(uq + 4);

            // u @ U^T : 16 chained f32 WMMAs (K=64)
            v8f acc_u = {};
#pragma unroll
            for (int s = 0; s < 16; ++s)
                acc_u = __builtin_amdgcn_wmma_f32_16x16x4_f32(
                    false, au[s], false, bu[s], (short)0, acc_u, false, false);

            // y @ Y^T : 4 chained bf16 WMMAs (K=128, exact for 0/1 data)
            v8f acc_y = {};
#pragma unroll
            for (int c = 0; c < 4; ++c)
                acc_y = __builtin_amdgcn_wmma_f32_16x16x32_bf16(
                    false, ay[c], false, by[c], (short)0, acc_y, false, false);

            // D[m][n]: lane holds n = idx, m = r + 8*hf for r=0..7
#pragma unroll
            for (int r = 0; r < 8; ++r) {
                const float usq_m = (r < 4) ? q0[r & 3] : q1[r & 3];
                float dist = usq_m - 2.0f * acc_u[r] + Usq_n;
                dist = fmaxf(dist, 0.0f);
                const bool mism = (acc_y[r] == 0.0f);
                lane_loss += mism ? 0.5f * fmaxf(MARGIN - dist, 0.0f)
                                  : 0.5f * dist;
            }
        }
    }

    // deterministic fixed-order block reduction
    __shared__ float red[128];
    red[threadIdx.x] = lane_loss;
    __syncthreads();
#pragma unroll
    for (int st = 64; st > 0; st >>= 1) {
        if (threadIdx.x < st) red[threadIdx.x] += red[threadIdx.x + st];
        __syncthreads();
    }
    if (threadIdx.x == 0) partials[blockIdx.x] = red[0];
}

// ---------------- finalize: sum partials (loss1) + sign regularizer (loss2)
__global__ __launch_bounds__(256) void dsh_finalize(
    const float* __restrict__ partials, const float* __restrict__ u,
    float* __restrict__ out)
{
    __shared__ float red[256];
    __shared__ float s_loss1;

    float s1 = 0.0f;
    for (int i = threadIdx.x; i < NBLOCKS; i += 256) s1 += partials[i];
    red[threadIdx.x] = s1;
    __syncthreads();
#pragma unroll
    for (int st = 128; st > 0; st >>= 1) {
        if (threadIdx.x < st) red[threadIdx.x] += red[threadIdx.x + st];
        __syncthreads();
    }
    if (threadIdx.x == 0)
        s_loss1 = red[0] / ((float)BATCH * (float)NTRAIN);
    __syncthreads();

    float s2 = 0.0f;
    for (int i = threadIdx.x; i < BATCH * BIT; i += 256) {
        float x = u[i];                 // |1 - sign(x)|: 0 if x>0, 2 if x<0, 1 if x==0
        s2 += (x > 0.0f) ? 0.0f : ((x < 0.0f) ? 2.0f : 1.0f);
    }
    red[threadIdx.x] = s2;
    __syncthreads();
#pragma unroll
    for (int st = 128; st > 0; st >>= 1) {
        if (threadIdx.x < st) red[threadIdx.x] += red[threadIdx.x + st];
        __syncthreads();
    }
    if (threadIdx.x == 0)
        out[0] = s_loss1 + ALPHA * (red[0] / (float)(BATCH * BIT));
}

// ---------------- launcher ----------------
extern "C" void kernel_launch(void* const* d_in, const int* in_sizes, int n_in,
                              void* d_out, int out_size, void* d_ws, size_t ws_size,
                              hipStream_t stream) {
    (void)in_sizes; (void)n_in; (void)out_size; (void)ws_size;
    const float* u   = (const float*)d_in[0];
    const float* y   = (const float*)d_in[1];
    const int*   ind = (const int*)d_in[2];
    const float* U   = (const float*)d_in[3];
    const float* Y   = (const float*)d_in[4];
    float* out = (float*)d_out;

    char* ws = (char*)d_ws;
    int*    ovr      = (int*)(ws + WS_OVR);
    float*  usq      = (float*)(ws + WS_USQ);
    float*  u_reord  = (float*)(ws + WS_UREORD);
    __bf16* ybf      = (__bf16*)(ws + WS_YBF);
    float*  partials = (float*)(ws + WS_PART);

    dsh_init_ovr<<<(NTRAIN + 255) / 256, 256, 0, stream>>>(ovr);
    dsh_set_ovr<<<(BATCH + 255) / 256, 256, 0, stream>>>(ovr, ind);
    dsh_pack_rows<<<(BATCH + 255) / 256, 256, 0, stream>>>(u, y, usq, u_reord, ybf);
    dsh_main<<<NBLOCKS, 128, 0, stream>>>(u, y, U, Y, ovr, usq, u_reord, ybf, partials);
    dsh_finalize<<<1, 256, 0, stream>>>(partials, u, out);
}
